// DyGPrompt_pretrain_9079560864230
// MI455X (gfx1250) — compile-verified, matrix operand order checked by reading, not run on hardware
//
#include <hip/hip_runtime.h>
#include <hip/hip_bf16.h>

typedef __attribute__((ext_vector_type(16))) __bf16 v16bf;
typedef __attribute__((ext_vector_type(4)))  __bf16 v4bf;
typedef __attribute__((ext_vector_type(2)))  __bf16 v2bf;
typedef __attribute__((ext_vector_type(8)))  float  v8f;

#define K_NB      20
#define E_DIM     172
#define DE_DIM    172
#define KV_DIM    344      // E + DE
#define KVP       352      // padded to 11*32
#define EP        176      // E padded to 11*16
#define QKP       192      // E padded to 6*32 (k-dim for q/out gemms)
#define HEADS     2
#define HD        86
#define S_PER_BLK 4
#define ROWS      (S_PER_BLK * K_NB)   // 80 kv rows = 5 m-tiles of 16
#define NNODES    100000
#define E4        43                   // 172 / 4

// ---------------- WMMA fragment loader ----------------
// CDNA5 16-bit A/B fragment: per lane, elements 0..7  = src[k0+kb .. k0+kb+7],
//                                      elements 8..15 = src[k0+kb+16 .. k0+kb+23]
// (kb = 0 for lanes 0-15, 8 for lanes 16-31).  Both chunks are 16B-aligned when
// pitch and k0 are multiples of 32 elements -> two b128 loads, no guards.
__device__ __forceinline__ v16bf load_frag(const __bf16* rowbase, int k0, int kb) {
  const __bf16* p = rowbase + k0 + kb;
  union { uint4 u[2]; v16bf v; } u;
  u.u[0] = *(const uint4*)(p);
  u.u[1] = *(const uint4*)(p + 16);
  return u.v;
}

// ---------------- weight pre-pack: padded bf16 tables + padded biases ----------------
__global__ void pack_weights_kernel(const float* __restrict__ qW, const float* __restrict__ kW,
                                    const float* __restrict__ vW, const float* __restrict__ outW,
                                    const float* __restrict__ bq, const float* __restrict__ bk,
                                    const float* __restrict__ bv, const float* __restrict__ ob,
                                    __bf16* __restrict__ qWp, __bf16* __restrict__ kWp,
                                    __bf16* __restrict__ vWp, __bf16* __restrict__ oWp,
                                    float* __restrict__ biasP) {
  int i = blockIdx.x * blockDim.x + threadIdx.x;
  const int nKV = EP * KVP;   // 61952
  const int nQ  = EP * QKP;   // 33792
  if (i < nKV) {
    int n = i / KVP, k = i - n * KVP;
    bool v = (n < E_DIM) && (k < KV_DIM);
    kWp[i] = (__bf16)(v ? kW[n * KV_DIM + k] : 0.0f);
    vWp[i] = (__bf16)(v ? vW[n * KV_DIM + k] : 0.0f);
  }
  if (i < nQ) {
    int n = i / QKP, k = i - n * QKP;
    bool v = (n < E_DIM) && (k < E_DIM);
    qWp[i] = (__bf16)(v ? qW[n * E_DIM + k] : 0.0f);
    oWp[i] = (__bf16)(v ? outW[n * E_DIM + k] : 0.0f);
  }
  if (i < 4 * EP) {
    int which = i / EP, n = i - which * EP;
    const float* b = (which == 0) ? bq : (which == 1) ? bk : (which == 2) ? bv : ob;
    biasP[i] = (n < E_DIM) ? b[n] : 0.0f;
  }
}

// ---------------- small helper kernels ----------------
__global__ void time_encode_kernel(const float* __restrict__ ts, const float* __restrict__ tw,
                                   const float* __restrict__ tb, float* __restrict__ t, int n) {
  int i = blockIdx.x * blockDim.x + threadIdx.x;
  if (i < n) {
    int b = i / E_DIM;
    int e = i - b * E_DIM;
    t[i] = cosf(ts[b] * tw[e] + tb[e]);
  }
}

__global__ void winner_init_kernel(int* __restrict__ w, int n) {
  int i = blockIdx.x * blockDim.x + threadIdx.x;
  if (i < n) w[i] = -1;
}

__global__ void winner_scatter_kernel(const int* __restrict__ nodes, int* __restrict__ w, int n) {
  int i = blockIdx.x * blockDim.x + threadIdx.x;
  if (i < n) atomicMax(&w[nodes[i]], i);   // deterministic: max duplicate wins
}

// ---------------- fused TGAT layer kernel ----------------
__global__ __launch_bounds__(256)
void tgat_layer_kernel(const int* __restrict__ nodes,
                       const int* __restrict__ neighbors,
                       const int* __restrict__ edge_idxs,
                       const float* __restrict__ nfeat,
                       const float* __restrict__ efeat,
                       const float* __restrict__ tEnc,
                       const __bf16* __restrict__ qWp, const __bf16* __restrict__ kWp,
                       const __bf16* __restrict__ vWp, const __bf16* __restrict__ oWp,
                       const float* __restrict__ biasP,
                       const float* __restrict__ hPrev,
                       const int* __restrict__ winner,
                       float* __restrict__ hOut,
                       int layer, int B) {
  __shared__ __align__(16) __bf16 kvLds[ROWS * KVP];        // 56320 B
  __shared__ __align__(16) __bf16 koLds[ROWS * EP];         // 28160 B
  __shared__ __align__(16) __bf16 voLds[ROWS * EP];         // 28160 B
  __shared__ __align__(16) __bf16 qinLds[16 * QKP];         //  6144 B
  __shared__ __align__(16) float  qLds[16 * EP];            // 11264 B
  __shared__ __align__(16) __bf16 aoLds[16 * QKP];          //  6144 B
  __shared__ __align__(16) float  tLds[S_PER_BLK * E_DIM];  //  2752 B
  __shared__ float  attnLds[S_PER_BLK * HEADS * K_NB];
  __shared__ int    nbLds[ROWS];
  __shared__ int    eiLds[ROWS];
  __shared__ const float* srcLds[ROWS];
  __shared__ const float* qsrcLds[S_PER_BLK];

  const int tid  = threadIdx.x;
  const int wave = tid >> 5;
  const int lane = tid & 31;
  const int ln   = lane & 15;
  const int kb   = (lane >> 4) << 3;          // fragment K sub-offset (0 or 8)
  const int mhi  = (lane >> 4) << 3;          // C/D fragment row offset (0 or 8)
  const int b0   = blockIdx.x * S_PER_BLK;

  // ---- stage 0: metadata, source pointers, zero pad regions ----
  if (tid < ROWS) {
    int s  = tid / K_NB;
    int k  = tid - s * K_NB;
    int b  = min(b0 + s, B - 1);
    int nb = neighbors[(size_t)b * K_NB + k];
    int ei = edge_idxs[(size_t)b * K_NB + k];
    nbLds[tid] = nb;
    eiLds[tid] = ei;
    const float* src;
    if (layer == 0) {
      src = nfeat + (size_t)nb * E_DIM;
    } else {
      int w = winner[nb];
      src = (w >= 0) ? (hPrev + (size_t)w * E_DIM) : (nfeat + (size_t)nb * E_DIM);
    }
    srcLds[tid] = src;
    __builtin_prefetch(src, 0, 0);                              // global_prefetch_b8
    __builtin_prefetch(efeat + (size_t)ei * DE_DIM, 0, 0);
    // zero kv pad columns 344..351 (16 bytes per row)
    uint4 z; z.x = z.y = z.z = z.w = 0u;
    *(uint4*)&kvLds[tid * KVP + KV_DIM] = z;
  }
  if (tid < S_PER_BLK) {
    int b  = min(b0 + tid, B - 1);
    int nd = nodes[b];
    const float* src;
    if (layer == 0) {
      src = nfeat + (size_t)nd * E_DIM;
    } else {
      int w = winner[nd];
      src = (w >= 0) ? (hPrev + (size_t)w * E_DIM) : (nfeat + (size_t)nd * E_DIM);
    }
    qsrcLds[tid] = src;
  }
  {
    uint4 z; z.x = z.y = z.z = z.w = 0u;
    uint4* qz = (uint4*)qinLds;
    uint4* az = (uint4*)aoLds;
    for (int i = tid; i < (16 * QKP) / 8; i += 256) { qz[i] = z; az[i] = z; }
  }
  for (int e = tid; e < S_PER_BLK * E_DIM; e += 256) {
    int s = e / E_DIM;
    int c = e - s * E_DIM;
    int b = min(b0 + s, B - 1);
    tLds[e] = tEnc[(size_t)b * E_DIM + c];
  }
  __syncthreads();

  // ---- stage 1: pack kv rows (bf16) and q_in (bf16), float4 granularity ----
  for (int e = tid; e < ROWS * E4; e += 256) {          // node-feature + t region
    int r  = e / E4;
    int c4 = e - r * E4;
    int s  = r / K_NB;
    float4 a = ((const float4*)srcLds[r])[c4];
    float4 t = ((const float4*)(tLds + s * E_DIM))[c4];
    v4bf o;
    o[0] = (__bf16)(a.x + t.x); o[1] = (__bf16)(a.y + t.y);
    o[2] = (__bf16)(a.z + t.z); o[3] = (__bf16)(a.w + t.w);
    *(v4bf*)&kvLds[r * KVP + (c4 << 2)] = o;
  }
  for (int e = tid; e < ROWS * E4; e += 256) {          // edge-feature region
    int r  = e / E4;
    int c4 = e - r * E4;
    float4 a = ((const float4*)(efeat + (size_t)eiLds[r] * DE_DIM))[c4];
    v4bf o;
    o[0] = (__bf16)a.x; o[1] = (__bf16)a.y; o[2] = (__bf16)a.z; o[3] = (__bf16)a.w;
    *(v4bf*)&kvLds[r * KVP + E_DIM + (c4 << 2)] = o;
  }
  for (int e = tid; e < S_PER_BLK * E4; e += 256) {     // q_in rows 0..3
    int r  = e / E4;
    int c4 = e - r * E4;
    float4 a = ((const float4*)qsrcLds[r])[c4];
    float4 t = ((const float4*)(tLds + r * E_DIM))[c4];
    v4bf o;
    o[0] = (__bf16)(a.x + t.x); o[1] = (__bf16)(a.y + t.y);
    o[2] = (__bf16)(a.z + t.z); o[3] = (__bf16)(a.w + t.w);
    *(v4bf*)&qinLds[r * QKP + (c4 << 2)] = o;
  }
  __syncthreads();

  // ---- stage 2+3: K/V projections (22 jobs) + q projection (11 jobs) ----
  for (int j = wave; j < 33; j += 8) {
    if (j < 22) {
      const bool    isK = (j < 11);
      const __bf16* W   = isK ? kWp : vWp;
      const float*  bia = biasP + (isK ? EP : 2 * EP);
      __bf16*       dst = isK ? koLds : voLds;
      int n0 = (isK ? j : j - 11) * 16;
      const __bf16* wrow = W + (size_t)(n0 + ln) * KVP;
      v8f acc[5] = {};
      for (int ks = 0; ks < 11; ++ks) {
        int k0 = ks * 32;
        v16bf bf = load_frag(wrow, k0, kb);
#pragma unroll
        for (int m = 0; m < 5; ++m) {
          v16bf af = load_frag(kvLds + (size_t)(m * 16 + ln) * KVP, k0, kb);
          acc[m] = __builtin_amdgcn_wmma_f32_16x16x32_bf16(false, af, false, bf,
                                                           (short)0, acc[m], false, false);
        }
      }
      int n = n0 + ln;
      float bb = bia[n];
#pragma unroll
      for (int m = 0; m < 5; ++m)
#pragma unroll
        for (int i = 0; i < 8; ++i)
          dst[(size_t)(m * 16 + mhi + i) * EP + n] = (__bf16)(acc[m][i] + bb);
    } else {
      int n0 = (j - 22) * 16;
      const __bf16* wrow = qWp + (size_t)(n0 + ln) * QKP;
      v8f acc = {};
      for (int ks = 0; ks < 6; ++ks) {
        int k0 = ks * 32;
        v16bf bf = load_frag(wrow, k0, kb);
        v16bf af = load_frag(qinLds + (size_t)ln * QKP, k0, kb);
        acc = __builtin_amdgcn_wmma_f32_16x16x32_bf16(false, af, false, bf,
                                                      (short)0, acc, false, false);
      }
      int n = n0 + ln;
      float bb = biasP[n];                   // bq
#pragma unroll
      for (int i = 0; i < 8; ++i)
        qLds[(size_t)(mhi + i) * EP + n] = acc[i] + bb;
    }
  }
  __syncthreads();

  // ---- stage 4: scores + masked softmax; one wave per (sample, head) ----
  {
    int s  = wave >> 1;
    int hh = wave & 1;
    float sc = -1.0e9f;
    if (lane < K_NB) {
      float d = 0.0f;
      const v2bf*  kr = (const v2bf*)&koLds[(size_t)(s * K_NB + lane) * EP + hh * HD];
      const float* qr = &qLds[(size_t)s * EP + hh * HD];
#pragma unroll
      for (int c = 0; c < HD / 2; ++c) {
        v2bf p = kr[c];
        d += (float)p[0] * qr[2 * c] + (float)p[1] * qr[2 * c + 1];
      }
      bool inval = true;
      for (int k = 0; k < K_NB; ++k)
        if (nbLds[s * K_NB + k] != 0) { inval = false; break; }
      bool pad = (nbLds[s * K_NB + lane] == 0) && !(inval && lane == 0);
      sc = pad ? -1.0e9f : d * 0.10783277320343841f;   // 1/sqrt(86)
    }
    float mx = sc;
    for (int off = 16; off > 0; off >>= 1) mx = fmaxf(mx, __shfl_xor(mx, off, 32));
    float ex = (lane < K_NB) ? __expf(sc - mx) : 0.0f;
    float sm = ex;
    for (int off = 16; off > 0; off >>= 1) sm += __shfl_xor(sm, off, 32);
    if (lane < K_NB) attnLds[(s * HEADS + hh) * K_NB + lane] = ex / sm;
  }
  __syncthreads();

  // ---- stage 5: attn-weighted sum of V -> aoLds ----
  for (int e = tid; e < S_PER_BLK * E_DIM; e += 256) {
    int s  = e / E_DIM;
    int c  = e - s * E_DIM;
    int hh = c / HD;
    const float* aw = &attnLds[(s * HEADS + hh) * K_NB];
    float d = 0.0f;
#pragma unroll 4
    for (int k = 0; k < K_NB; ++k)
      d += aw[k] * (float)voLds[(size_t)(s * K_NB + k) * EP + c];
    aoLds[s * QKP + c] = (__bf16)d;
  }
  __syncthreads();

  // ---- stage 6: output projection -> hOut ----
  for (int nt = wave; nt < 11; nt += 8) {
    int n0 = nt * 16;
    const __bf16* wrow = oWp + (size_t)(n0 + ln) * QKP;
    v8f acc = {};
    for (int ks = 0; ks < 6; ++ks) {
      int k0 = ks * 32;
      v16bf bf = load_frag(wrow, k0, kb);
      v16bf af = load_frag(aoLds + (size_t)ln * QKP, k0, kb);
      acc = __builtin_amdgcn_wmma_f32_16x16x32_bf16(false, af, false, bf,
                                                    (short)0, acc, false, false);
    }
    int n = n0 + ln;
    if (n < E_DIM) {
      float bb = biasP[3 * EP + n];
#pragma unroll
      for (int i = 0; i < 8; ++i) {
        int m = mhi + i;
        if (m < S_PER_BLK && (b0 + m) < B)
          hOut[(size_t)(b0 + m) * E_DIM + n] = acc[i] + bb;
      }
    }
  }
}

// ---------------- host-side launch ----------------
static inline char* align256(char* p) {
  return (char*)(((uintptr_t)p + 255) & ~(uintptr_t)255);
}

extern "C" void kernel_launch(void* const* d_in, const int* in_sizes, int n_in,
                              void* d_out, int out_size, void* d_ws, size_t ws_size,
                              hipStream_t stream) {
  const int*   nodes     = (const int*)d_in[0];
  const float* ts        = (const float*)d_in[1];
  const int*   neighbors = (const int*)d_in[2];
  const int*   eidx      = (const int*)d_in[3];
  const float* nfeat     = (const float*)d_in[4];
  const float* efeat     = (const float*)d_in[5];
  const float* tw        = (const float*)d_in[6];
  const float* tb        = (const float*)d_in[7];
  const float* qW        = (const float*)d_in[8];
  const float* kW        = (const float*)d_in[9];
  const float* vW        = (const float*)d_in[10];
  const float* bq        = (const float*)d_in[11];
  const float* bk        = (const float*)d_in[12];
  const float* bv        = (const float*)d_in[13];
  const float* outW      = (const float*)d_in[14];
  const float* outB      = (const float*)d_in[15];
  float* out = (float*)d_out;

  const int B = in_sizes[0];                 // 12000

  char* ws = (char*)d_ws;
  float*  tEnc = (float*)ws;   ws = align256(ws + (size_t)B * E_DIM * sizeof(float));
  float*  h0   = (float*)ws;   ws = align256(ws + (size_t)B * E_DIM * sizeof(float));
  int*    win  = (int*)ws;     ws = align256(ws + (size_t)NNODES * sizeof(int));
  __bf16* kWp  = (__bf16*)ws;  ws = align256(ws + (size_t)EP * KVP * sizeof(__bf16));
  __bf16* vWp  = (__bf16*)ws;  ws = align256(ws + (size_t)EP * KVP * sizeof(__bf16));
  __bf16* qWp  = (__bf16*)ws;  ws = align256(ws + (size_t)EP * QKP * sizeof(__bf16));
  __bf16* oWp  = (__bf16*)ws;  ws = align256(ws + (size_t)EP * QKP * sizeof(__bf16));
  float*  bP   = (float*)ws;   ws = align256(ws + (size_t)4 * EP * sizeof(float));

  const int nPack = EP * KVP;  // largest pack domain
  pack_weights_kernel<<<(nPack + 255) / 256, 256, 0, stream>>>(
      qW, kW, vW, outW, bq, bk, bv, outB, qWp, kWp, vWp, oWp, bP);

  const int nT = B * E_DIM;
  time_encode_kernel<<<(nT + 255) / 256, 256, 0, stream>>>(ts, tw, tb, tEnc, nT);

  const int nBlocks = (B + S_PER_BLK - 1) / S_PER_BLK;

  // layer 0: sources are node_features; output -> h0
  tgat_layer_kernel<<<nBlocks, 256, 0, stream>>>(
      nodes, neighbors, eidx, nfeat, efeat, tEnc,
      qWp, kWp, vWp, oWp, bP, h0, win, h0, /*layer=*/0, B);

  // deterministic duplicate resolution for nf.at[nodes].set(h)
  winner_init_kernel<<<(NNODES + 255) / 256, 256, 0, stream>>>(win, NNODES);
  winner_scatter_kernel<<<(B + 255) / 256, 256, 0, stream>>>(nodes, win, B);

  // layer 1: sources override via winner/h0; output -> d_out (== h flat)
  tgat_layer_kernel<<<nBlocks, 256, 0, stream>>>(
      nodes, neighbors, eidx, nfeat, efeat, tEnc,
      qWp, kWp, vWp, oWp, bP, h0, win, out, /*layer=*/1, B);
}
